// GraphSAGEDecoder_18167711662114
// MI455X (gfx1250) — matrix operand order repaired and checked by reading, not compiled, not run on hardware
//
#include <hip/hip_runtime.h>
#include <hip/hip_bf16.h>
#include <math.h>

// ---------------------------------------------------------------------------
// GraphSAGE decoder for MI455X (gfx1250, wave32).
// Big N*N GEMMs run on v_wmma_f32_16x16x32_bf16 (f32 accumulate), operands
// software-pipelined through double-buffered LDS as packed bf16 K-pairs.
// K loop manually unrolled 2x so buffer offsets are immediates; edge-mask
// handling peeled out of the steady state; B staging sized so every thread
// owns an exact unit (no exec-masked loads).
// ---------------------------------------------------------------------------

typedef __attribute__((ext_vector_type(16))) __bf16 v16bf;
typedef __attribute__((ext_vector_type(8)))  float  v8f;

#define NN 2048
#define BATCH 4

union FragBF {
    unsigned int u[8];
    v16bf        v;
};

union BF2Bits {
    __hip_bfloat162 h;
    unsigned int    u;
};

__device__ __forceinline__ unsigned int bfpack(float lo, float hi) {
    BF2Bits t;
    t.h = __float22bfloat162_rn(make_float2(lo, hi));  // v_cvt_pk_bf16_f32
    return t.u;                                        // lo in [15:0], hi in [31:16]
}

// ---------------------------------------------------------------------------
// C[b, i, f] = sum_k Aop[b, i, k] * Bm[b, k, f]
//   TRANSA=true : Aop[i,k] = adj[b,k,i] * (k<N-2 && i<N-2 ? wadj[k,i] : 1)
//   TRANSA=false: Aop[i,k] = inv_deg[b,i,k]
// Workgroup: 256 threads (8 waves), 64 x F output tile, K staged 32 at a time,
// double-buffered LDS, one barrier per K-step.
// ---------------------------------------------------------------------------
template <int F, bool TRANSA>
__global__ __launch_bounds__(256)
void sage_gemm_wmma_bf16(const float* __restrict__ A,
                         const float* __restrict__ WADJ,
                         const float* __restrict__ Bm,
                         float* __restrict__ C)
{
    constexpr int MT  = 64;
    constexpr int KT  = 32;
    constexpr int LDA = KT + 4;           // 36 halfwords = 18 dwords; uint2 stores stay 8B-aligned
    constexpr int CT  = F / 32;           // column tiles per wave
    constexpr int FV  = (F == 32) ? 2 : 4;    // features per B staging unit
    constexpr int BU  = F / (16 * FV);        // B units per thread: exactly 1 or 2, no remainder

    __shared__ __align__(16) unsigned short lA0[MT * LDA];
    __shared__ __align__(16) unsigned short lA1[MT * LDA];
    __shared__ __align__(16) unsigned short lB0[F * LDA];
    __shared__ __align__(16) unsigned short lB1[F * LDA];

    const int tid  = threadIdx.x;
    const int wave = tid >> 5;
    const int lane = tid & 31;
    const int b    = blockIdx.y;
    const int rowBase = blockIdx.x * MT;

    const size_t batA = (size_t)b * NN * NN;
    const size_t batB = (size_t)b * NN * F;

    const int rowTile = wave & 3;          // 0..3 -> 16-row sub-tile
    const int colGrp  = wave >> 2;         // 0..1 -> interleaved column tiles
    const int mloc    = rowTile * 16 + (lane & 15);
    const int kbA     = (lane >> 4) * 16;  // A frag K-base (halves)
    const int kbB     = (lane >> 4) * 16;  // B frag K-base

    // staging coordinates (thread-invariant)
    const int aM0 = (tid & 15) * 4;        // TRANSA: 4 consecutive i
    const int aK  = (tid >> 4) * 2;        // TRANSA: K pair
    const int rM  = tid >> 3;              // row-major: m and m+32
    const int rKK = (tid & 7) * 4;         // row-major: 4 consecutive k

    const bool rowInterior = (rowBase + MT) <= (NN - 2);

    // ----- register staging for the software pipeline -----
    float4 aR0, aR1, wR0, wR1;             // TRANSA operands
    float4 rR0, rR1;                       // row-major operands
    float  bS0[BU][FV], bS1[BU][FV];       // B operands (two K rows)

    auto loadTile = [&](int ks) {
        if (TRANSA) {
            const int gi0 = rowBase + aM0;
            const int gk  = ks + aK;
            aR0 = *(const float4*)&A[batA + (size_t)gk * NN + gi0];
            aR1 = *(const float4*)&A[batA + (size_t)(gk + 1) * NN + gi0];
            wR0 = *(const float4*)&WADJ[(size_t)gk * NN + gi0];
            wR1 = *(const float4*)&WADJ[(size_t)(gk + 1) * NN + gi0];
        } else {
            rR0 = *(const float4*)&A[batA + (size_t)(rowBase + rM) * NN + (ks + rKK)];
            rR1 = *(const float4*)&A[batA + (size_t)(rowBase + rM + 32) * NN + (ks + rKK)];
        }
#pragma unroll
        for (int e = 0; e < BU; ++e) {
            const int idx = tid + e * 256;
            const int f0  = (idx % (F / FV)) * FV;
            const int kk  = (idx / (F / FV)) * 2;
            const float* p0 = &Bm[batB + (size_t)(ks + kk) * F + f0];
            const float* p1 = &Bm[batB + (size_t)(ks + kk + 1) * F + f0];
            if (FV == 4) {
                const float4 t0 = *(const float4*)p0;
                const float4 t1 = *(const float4*)p1;
                bS0[e][0] = t0.x; bS0[e][1] = t0.y; bS0[e][2] = t0.z; bS0[e][3] = t0.w;
                bS1[e][0] = t1.x; bS1[e][1] = t1.y; bS1[e][2] = t1.z; bS1[e][3] = t1.w;
            } else {
                const float2 t0 = *(const float2*)p0;
                const float2 t1 = *(const float2*)p1;
                bS0[e][0] = t0.x; bS0[e][1] = t0.y;
                bS1[e][0] = t1.x; bS1[e][1] = t1.y;
            }
        }
    };

    // kInt is a call-site literal: folds the edge path away in the main loop.
    auto storeTile = [&](int ks, unsigned short* A_, unsigned short* B_, bool kInt) {
        if (TRANSA) {
            const int gi0 = rowBase + aM0;
            const int gk  = ks + aK;
            if (kInt && rowInterior) {
                // interior fast path: mask all-ones, eff = wadj
                *(unsigned int*)&A_[(aM0 + 0) * LDA + aK] = bfpack(aR0.x * wR0.x, aR1.x * wR1.x);
                *(unsigned int*)&A_[(aM0 + 1) * LDA + aK] = bfpack(aR0.y * wR0.y, aR1.y * wR1.y);
                *(unsigned int*)&A_[(aM0 + 2) * LDA + aK] = bfpack(aR0.z * wR0.z, aR1.z * wR1.z);
                *(unsigned int*)&A_[(aM0 + 3) * LDA + aK] = bfpack(aR0.w * wR0.w, aR1.w * wR1.w);
            } else {
                // edge path: branchless per-element select (loads already in regs)
                const bool k0in = gk < (NN - 2), k1in = (gk + 1) < (NN - 2);
                const float e0x = (k0in && (gi0 + 0) < NN - 2) ? wR0.x : 1.0f;
                const float e0y = (k0in && (gi0 + 1) < NN - 2) ? wR0.y : 1.0f;
                const float e0z = (k0in && (gi0 + 2) < NN - 2) ? wR0.z : 1.0f;
                const float e0w = (k0in && (gi0 + 3) < NN - 2) ? wR0.w : 1.0f;
                const float e1x = (k1in && (gi0 + 0) < NN - 2) ? wR1.x : 1.0f;
                const float e1y = (k1in && (gi0 + 1) < NN - 2) ? wR1.y : 1.0f;
                const float e1z = (k1in && (gi0 + 2) < NN - 2) ? wR1.z : 1.0f;
                const float e1w = (k1in && (gi0 + 3) < NN - 2) ? wR1.w : 1.0f;
                *(unsigned int*)&A_[(aM0 + 0) * LDA + aK] = bfpack(aR0.x * e0x, aR1.x * e1x);
                *(unsigned int*)&A_[(aM0 + 1) * LDA + aK] = bfpack(aR0.y * e0y, aR1.y * e1y);
                *(unsigned int*)&A_[(aM0 + 2) * LDA + aK] = bfpack(aR0.z * e0z, aR1.z * e1z);
                *(unsigned int*)&A_[(aM0 + 3) * LDA + aK] = bfpack(aR0.w * e0w, aR1.w * e1w);
            }
        } else {
            uint2 p0, p1;
            p0.x = bfpack(rR0.x, rR0.y);
            p0.y = bfpack(rR0.z, rR0.w);
            p1.x = bfpack(rR1.x, rR1.y);
            p1.y = bfpack(rR1.z, rR1.w);
            *(uint2*)&A_[rM * LDA + rKK]        = p0;   // 8B aligned: 72*m + 2*rKK
            *(uint2*)&A_[(rM + 32) * LDA + rKK] = p1;
        }
#pragma unroll
        for (int e = 0; e < BU; ++e) {
            const int idx = tid + e * 256;
            const int f0  = (idx % (F / FV)) * FV;
            const int kk  = (idx / (F / FV)) * 2;
#pragma unroll
            for (int j = 0; j < FV; ++j) {
                *(unsigned int*)&B_[(f0 + j) * LDA + kk] = bfpack(bS0[e][j], bS1[e][j]);
            }
        }
    };

    v8f acc[CT];
#pragma unroll
    for (int j = 0; j < CT; ++j) acc[j] = v8f{};

    auto compute = [&](const unsigned short* A_, const unsigned short* B_) {
        FragBF fa;
        const unsigned int* pa = (const unsigned int*)&A_[mloc * LDA + kbA];
#pragma unroll
        for (int v = 0; v < 4; ++v) { fa.u[v] = pa[v]; fa.u[4 + v] = pa[4 + v]; }
#pragma unroll
        for (int j = 0; j < CT; ++j) {
            const int c    = colGrp + 2 * j;
            const int floc = c * 16 + (lane & 15);
            FragBF fb;
            const unsigned int* pb = (const unsigned int*)&B_[floc * LDA + kbB];
#pragma unroll
            for (int v = 0; v < 8; ++v) fb.u[v] = pb[v];
            acc[j] = __builtin_amdgcn_wmma_f32_16x16x32_bf16(
                false, fa.v, false, fb.v, (short)0, acc[j], false, false);
        }
    };

    // ----- prologue: stage K-tile 0 into buffer 0 -----
    loadTile(0);
    storeTile(0, lA0, lB0, true);

    // ----- steady state: all staged tiles are K-interior (ks <= 1984) -----
    int k0 = 0;
    for (; k0 < NN - 2 * KT; k0 += 2 * KT) {
        __syncthreads();
        loadTile(k0 + KT);
        compute(lA0, lB0);
        storeTile(k0 + KT, lA1, lB1, true);
        __syncthreads();
        loadTile(k0 + 2 * KT);
        compute(lA1, lB1);
        storeTile(k0 + 2 * KT, lA0, lB0, true);
    }

    // ----- peeled tail: k0 == NN - 2*KT; staged tile ks = 2016 hits the mask edge
    __syncthreads();
    loadTile(k0 + KT);
    compute(lA0, lB0);                 // tile k = 1984
    storeTile(k0 + KT, lA1, lB1, false);
    __syncthreads();
    compute(lA1, lB1);                 // tile k = 2016

    // -------- write 16x16 f32 tiles: VGPR r -> M = r (+8 for hi half) -----
#pragma unroll
    for (int j = 0; j < CT; ++j) {
        const int c  = colGrp + 2 * j;
        const int f  = c * 16 + (lane & 15);
        const int mb = rowTile * 16 + (lane >> 4) * 8;
#pragma unroll
        for (int r = 0; r < 8; ++r) {
            C[batB + (size_t)(rowBase + mb + r) * F + f] = acc[j][r];
        }
    }
}

// ---------------------------------------------------------------------------
// Epilogue per layer: c = [H @ W, X @ Bp], L2-normalize rows, leaky-ReLU.
// One wave per node row.
// ---------------------------------------------------------------------------
template <int FIN, int FOUT>
__global__ __launch_bounds__(256)
void sage_tail(const float* __restrict__ H, const float* __restrict__ X,
               const float* __restrict__ W, const float* __restrict__ Bp,
               float* __restrict__ Xn)
{
    const int wave = threadIdx.x >> 5;
    const int lane = threadIdx.x & 31;
    const int row  = blockIdx.x * 8 + wave;     // flat (b*N + i)
    const float* h = H + (size_t)row * FIN;
    const float* x = X + (size_t)row * FIN;

    constexpr int T = (2 * FOUT) / 32;
    float val[T];
    float ssq = 0.f;
#pragma unroll
    for (int t = 0; t < T; ++t) {
        int col = lane + t * 32;
        float s = 0.f;
        if (col < FOUT) {
            for (int q = 0; q < FIN; ++q) s += h[q] * W[q * FOUT + col];
        } else {
            int c2 = col - FOUT;
            for (int q = 0; q < FIN; ++q) s += x[q] * Bp[q * FOUT + c2];
        }
        val[t] = s;
        ssq += s * s;
    }
#pragma unroll
    for (int m = 16; m >= 1; m >>= 1) ssq += __shfl_xor(ssq, m, 32);
    float nrm = fmaxf(sqrtf(ssq), 1e-12f);
    float inv = 1.0f / nrm;
#pragma unroll
    for (int t = 0; t < T; ++t) {
        int col = lane + t * 32;
        float y = val[t] * inv;
        y = (y >= 0.f) ? y : 0.1f * y;          // leaky ReLU (SLOPE = 0.1)
        Xn[(size_t)row * (2 * FOUT) + col] = y;
    }
}

// ---------------------------------------------------------------------------
// Decoder: a = X3[b, N-2], bb = X3[b, N-1]; t = a P1 P2 P1^T; y = <t, bb>.
// Single workgroup, 64 threads per batch.
// ---------------------------------------------------------------------------
__global__ __launch_bounds__(256)
void sage_decode(const float* __restrict__ X3,  // (B, N, 64)
                 const float* __restrict__ P1,  // (64, 32)
                 const float* __restrict__ P2,  // (32, 32)
                 float* __restrict__ out)       // (B, 1)
{
    __shared__ float u[BATCH][32];
    __shared__ float v[BATCH][32];
    __shared__ float red[BATCH][64];

    const int tid = threadIdx.x;
    const int b   = tid >> 6;
    const int j   = tid & 63;

    const float* a  = X3 + ((size_t)b * NN + (NN - 2)) * 64;
    const float* bb = X3 + ((size_t)b * NN + (NN - 1)) * 64;

    if (j < 32) {
        float s = 0.f;
        for (int i = 0; i < 64; ++i) s += a[i] * P1[i * 32 + j];
        u[b][j] = s;
    }
    __syncthreads();
    if (j < 32) {
        float s = 0.f;
        for (int k = 0; k < 32; ++k) s += u[b][k] * P2[k * 32 + j];
        v[b][j] = s;
    }
    __syncthreads();
    float t = 0.f;
    for (int k = 0; k < 32; ++k) t += v[b][k] * P1[j * 32 + k];
    red[b][j] = t * bb[j];
    __syncthreads();
    if (j == 0) {
        float s = 0.f;
        for (int i = 0; i < 64; ++i) s += red[b][i];
        out[b] = s;
    }
}

// ---------------------------------------------------------------------------
extern "C" void kernel_launch(void* const* d_in, const int* in_sizes, int n_in,
                              void* d_out, int out_size, void* d_ws, size_t ws_size,
                              hipStream_t stream)
{
    const float* x       = (const float*)d_in[0];   // (4, 2048, 32)
    const float* adj     = (const float*)d_in[1];   // (4, 2048, 2048)
    const float* inv_deg = (const float*)d_in[2];   // (4, 2048, 2048)
    const float* wadj1   = (const float*)d_in[3];   // (2048, 2048)
    const float* W1      = (const float*)d_in[4];   // (32, 32)
    const float* b1      = (const float*)d_in[5];   // (32, 32)
    const float* wadj2   = (const float*)d_in[6];
    const float* W2      = (const float*)d_in[7];   // (64, 64)
    const float* b2      = (const float*)d_in[8];
    const float* wadj3   = (const float*)d_in[9];
    const float* W3      = (const float*)d_in[10];  // (128, 32)
    const float* b3      = (const float*)d_in[11];
    const float* P1      = (const float*)d_in[12];  // (64, 32)
    const float* P2      = (const float*)d_in[13];  // (32, 32)
    float*       out     = (float*)d_out;           // (4, 1)

    // workspace layout (floats): T(1M) H(1M) X1(512K) X2(1M) X3(512K) = 16 MB
    float* ws = (float*)d_ws;
    float* T  = ws;
    float* H  = ws + 1048576;
    float* X1 = ws + 2097152;
    float* X2 = ws + 2621440;
    float* X3 = ws + 3670016;

    dim3 gg(NN / 64, BATCH);
    dim3 bt(256);
    const int tailGrid = BATCH * NN / 8;

    // ---- layer 1 (Fin = 32 -> c = 64) ----
    sage_gemm_wmma_bf16<32, true ><<<gg, bt, 0, stream>>>(adj,     wadj1, x, T);
    sage_gemm_wmma_bf16<32, false><<<gg, bt, 0, stream>>>(inv_deg, nullptr, T, H);
    sage_tail<32, 32><<<tailGrid, bt, 0, stream>>>(H, x, W1, b1, X1);

    // ---- layer 2 (Fin = 64 -> c = 128) ----
    sage_gemm_wmma_bf16<64, true ><<<gg, bt, 0, stream>>>(adj,     wadj2, X1, T);
    sage_gemm_wmma_bf16<64, false><<<gg, bt, 0, stream>>>(inv_deg, nullptr, T, H);
    sage_tail<64, 64><<<tailGrid, bt, 0, stream>>>(H, X1, W2, b2, X2);

    // ---- layer 3 (Fin = 128 -> c = 64) ----
    sage_gemm_wmma_bf16<128, true ><<<gg, bt, 0, stream>>>(adj,     wadj3, X2, T);
    sage_gemm_wmma_bf16<128, false><<<gg, bt, 0, stream>>>(inv_deg, nullptr, T, H);
    sage_tail<128, 32><<<tailGrid, bt, 0, stream>>>(H, X2, W3, b3, X3);

    // ---- decoder ----
    sage_decode<<<1, bt, 0, stream>>>(X3, P1, P2, out);
}
